// SAG_GAT_33663953666528
// MI455X (gfx1250) — matrix operand order repaired and checked by reading, not compiled, not run on hardware
//
#include <hip/hip_runtime.h>
#include <hip/hip_bf16.h>

typedef __attribute__((ext_vector_type(16))) __bf16 v16bf;
typedef __attribute__((ext_vector_type(8)))  __bf16 v8bf;
typedef __attribute__((ext_vector_type(8)))  float  v8f;

// ---------------- problem constants ----------------
namespace {
constexpr int N_  = 20000;
constexpr int E_  = 320000;
constexpr int DIN = 256;
constexpr int HH  = 512;   // HEADS*HID
constexpr int HID = 64;
constexpr int K1_ = 15000;
constexpr int K2_ = 11250;
constexpr int E1_ = E_ + N_;    // gat1 edges incl self loops
constexpr int E2_ = E_ + K1_;   // pooled-graph edges incl self loops

constexpr size_t alup(size_t x){ return (x + 255) & ~(size_t)255; }

// ---------------- workspace layout (bytes) ----------------
constexpr size_t OFF_XB   = 0;                                         // bf16 x       [N,256]
constexpr size_t OFF_W1T  = OFF_XB   + alup((size_t)N_*DIN*2);         // bf16 wlT,wrT [512][256] x2
constexpr size_t OFF_XL1  = OFF_W1T  + alup((size_t)2*HH*DIN*2);       // f32 xl       [N,512]  (later: xp [K1,512])
constexpr size_t OFF_XR1  = OFF_XL1  + alup((size_t)N_*HH*4);          // f32 xr       [N,512]  (later: gat2 region)
constexpr size_t OFF_AGG1 = OFF_XR1  + alup((size_t)N_*HH*4);          // f32 agg/h1   [N,512]
constexpr size_t OFF_ESC  = OFF_AGG1 + alup((size_t)N_*HH*4);          // edge scores  [(E+N)*8] f32 (reused)
constexpr size_t OFF_MKEY = OFF_ESC  + alup((size_t)E1_*8*4);          // u32 max keys [N*8]
constexpr size_t OFF_DEN  = OFF_MKEY + alup((size_t)N_*8*4);           // f32 denom    [N*8]
constexpr size_t OFF_NODE = OFF_DEN  + alup((size_t)N_*8*4);           // 6 node-scalar arrays [N]
constexpr size_t OFF_STATE= OFF_NODE + alup((size_t)6*N_*4);           // radix-select state (4KB)
constexpr size_t OFF_PERM = OFF_STATE+ 4096;                           // perm,vals,newid,perm2,vals2
constexpr size_t OFF_S1   = OFF_PERM + alup((size_t)(2*K1_ + N_ + 2*K2_)*4);
constexpr size_t OFF_MISC = OFF_S1   + alup((size_t)3*E_*4);           // small vectors
constexpr size_t WS_TOTAL = OFF_MISC + 65536;

// aliases inside region D (old xr1, 40.96MB)
constexpr size_t OFF_XPB  = OFF_XR1;                                   // bf16 xp      [K1,512]
constexpr size_t OFF_XL2  = OFF_XPB  + alup((size_t)K1_*HH*2);         // f32 xl2      [K1,64]
constexpr size_t OFF_XR2  = OFF_XL2  + alup((size_t)K1_*HID*4);        // f32 xr2
constexpr size_t OFF_AGG2 = OFF_XR2  + alup((size_t)K1_*HID*4);        // f32 agg2/h2
constexpr size_t OFF_W2T  = OFF_AGG2 + alup((size_t)K1_*HID*4);        // bf16 w2lT,w2rT [64][512] x2
// alias inside region C (old xl1)
constexpr size_t OFF_XP   = OFF_XL1;

static_assert(OFF_W2T + alup((size_t)2*HID*HH*2) <= OFF_AGG1, "region D overflow");
static_assert(OFF_XP + (size_t)K1_*HH*4 <= OFF_XR1, "region C overflow");
} // namespace

// ---------------- helpers ----------------
__device__ __forceinline__ unsigned fkey(float f){
  unsigned u = __float_as_uint(f);
  return (u & 0x80000000u) ? ~u : (u | 0x80000000u);
}
__device__ __forceinline__ float funkey(unsigned u){
  unsigned b = (u & 0x80000000u) ? (u & 0x7FFFFFFFu) : ~u;
  return __uint_as_float(b);
}
__device__ __forceinline__ float lrelu(float v){ return v > 0.f ? v : 0.2f*v; }

// ---------------- conversion kernels ----------------
__global__ void k_f32_to_bf16(const float* __restrict__ in, __bf16* __restrict__ out, long n){
  long i = (long)blockIdx.x*blockDim.x + threadIdx.x;
  if (i < n) out[i] = (__bf16)in[i];
}
// w [K][Nn] row-major -> wT [Nn][K] bf16
__global__ void k_transpose_bf16(const float* __restrict__ w, __bf16* __restrict__ wT, int K, int Nn){
  long i = (long)blockIdx.x*blockDim.x + threadIdx.x;
  if (i < (long)K*Nn){
    int r = (int)(i / Nn), c = (int)(i % Nn);
    wT[(size_t)c*K + r] = (__bf16)w[i];
  }
}

// ---------------- WMMA GEMM: C[M,N](f32) = A[M,K](bf16,row) * Bt[N,K](bf16,row of B^T) ----------------
// one wave per 16x16 C tile, K-loop step 32, v_wmma_f32_16x16x32_bf16
__global__ __launch_bounds__(256)
void k_wmma_gemm(const __bf16* __restrict__ A, const __bf16* __restrict__ Bt,
                 float* __restrict__ C, int M, int Nn, int K){
  int lane = threadIdx.x & 31;
  int tile = blockIdx.x * (blockDim.x >> 5) + (threadIdx.x >> 5);
  int tilesN = Nn >> 4;
  int tilesM = (M + 15) >> 4;
  if (tile >= tilesM * tilesN) return;
  int tm = tile / tilesN, tn = tile % tilesN;
  int half = lane >> 4;          // 0|1
  int l16  = lane & 15;
  int arow = tm*16 + l16; if (arow >= M) arow = M - 1;  // clamp; store predicated
  int bcol = tn*16 + l16;
  const __bf16* Ap = A  + (size_t)arow * K;
  const __bf16* Bp = Bt + (size_t)bcol * K;
  v8f acc = {0.f,0.f,0.f,0.f,0.f,0.f,0.f,0.f};
  for (int k = 0; k < K; k += 32){
    // A frag (16x32): elems 0..7 -> K=k+half*8+e ; elems 8..15 -> K=k+16+half*8+e
    v8bf a0 = *(const v8bf*)(Ap + k + half*8);
    v8bf a1 = *(const v8bf*)(Ap + k + 16 + half*8);
    // B frag (32x16, col l16): elems 0..7 -> K=k+half*8+e ; elems 8..15 -> K=k+16+half*8+e
    v8bf b0 = *(const v8bf*)(Bp + k + half*8);
    v8bf b1 = *(const v8bf*)(Bp + k + 16 + half*8);
    v16bf av, bv;
    #pragma unroll
    for (int i = 0; i < 8; ++i){ av[i]=a0[i]; av[i+8]=a1[i]; bv[i]=b0[i]; bv[i+8]=b1[i]; }
    acc = __builtin_amdgcn_wmma_f32_16x16x32_bf16(false, av, false, bv, (short)0, acc, false, false);
  }
  int rbase = tm*16 + half*8;  // C: lanes0-15 rows 0..7, lanes16-31 rows 8..15
  int ccol  = tn*16 + l16;
  #pragma unroll
  for (int i = 0; i < 8; ++i){
    int r = rbase + i;
    if (r < M) C[(size_t)r*Nn + ccol] = acc[i];
  }
}

// ---------------- gat1 (HEADS=8, C=64) edge passes: wave per edge ----------------
__global__ void k_gat1_p1(const int* __restrict__ src, const int* __restrict__ dst,
                          const float* __restrict__ xl, const float* __restrict__ xr,
                          const float* __restrict__ att, float* __restrict__ esc,
                          unsigned* __restrict__ mkey, int E, int Nn){
  int gid = blockIdx.x*blockDim.x + threadIdx.x;
  int e = gid >> 5, lane = gid & 31;
  if (e >= E + Nn) return;
  int s, d;
  if (e < E){ s = src[e]; d = dst[e]; } else { s = d = e - E; }
  const float* pl = xl + (size_t)s*HH + lane*16;
  const float* pr = xr + (size_t)d*HH + lane*16;
  const float* pa = att + lane*16;
  float acc = 0.f;
  #pragma unroll
  for (int i = 0; i < 16; ++i) acc += lrelu(pl[i] + pr[i]) * pa[i];
  acc += __shfl_xor(acc, 1, 32);
  acc += __shfl_xor(acc, 2, 32);
  if ((lane & 3) == 0){
    int h = lane >> 2;
    esc[(size_t)e*8 + h] = acc;
    atomicMax(&mkey[(size_t)d*8 + h], fkey(acc));
  }
}
__global__ void k_gat1_p2(const int* __restrict__ dst, float* __restrict__ esc,
                          const unsigned* __restrict__ mkey, float* __restrict__ den, int E, int Nn){
  int e = blockIdx.x*blockDim.x + threadIdx.x;
  if (e >= E + Nn) return;
  int d = (e < E) ? dst[e] : (e - E);
  #pragma unroll
  for (int h = 0; h < 8; ++h){
    float m  = funkey(mkey[(size_t)d*8 + h]);
    float ex = __expf(esc[(size_t)e*8 + h] - m);
    esc[(size_t)e*8 + h] = ex;
    atomicAdd(&den[(size_t)d*8 + h], ex);
  }
}
__global__ void k_gat1_p3(const int* __restrict__ src, const int* __restrict__ dst,
                          const float* __restrict__ xl, const float* __restrict__ esc,
                          const float* __restrict__ den, float* __restrict__ agg, int E, int Nn){
  int gid = blockIdx.x*blockDim.x + threadIdx.x;
  int e = gid >> 5, lane = gid & 31;
  if (e >= E + Nn) return;
  int s, d;
  if (e < E){ s = src[e]; d = dst[e]; } else { s = d = e - E; }
  int h = lane >> 2;
  float a = esc[(size_t)e*8 + h] / den[(size_t)d*8 + h];
  const float* pl = xl + (size_t)s*HH + lane*16;
  float* po = agg + (size_t)d*HH + lane*16;
  #pragma unroll
  for (int i = 0; i < 16; ++i) atomicAdd(&po[i], a * pl[i]);
}

// ---------------- gat2 (1 head, C=64) edge passes ----------------
__global__ void k_gat2_p1(const int* __restrict__ s1, const int* __restrict__ d1, const int* __restrict__ v1,
                          const float* __restrict__ xl, const float* __restrict__ xr,
                          const float* __restrict__ att, float* __restrict__ es,
                          unsigned* __restrict__ mkey, int E, int Kn){
  int gid = blockIdx.x*blockDim.x + threadIdx.x;
  int e = gid >> 5, lane = gid & 31;
  if (e >= E + Kn) return;
  int s, d;
  if (e < E){ if (!v1[e]) return; s = s1[e]; d = d1[e]; } else { s = d = e - E; }
  const float* pl = xl + (size_t)s*HID + lane*2;
  const float* pr = xr + (size_t)d*HID + lane*2;
  const float* pa = att + lane*2;
  float acc = lrelu(pl[0]+pr[0])*pa[0] + lrelu(pl[1]+pr[1])*pa[1];
  #pragma unroll
  for (int o = 16; o > 0; o >>= 1) acc += __shfl_xor(acc, o, 32);
  if (lane == 0){ es[e] = acc; atomicMax(&mkey[d], fkey(acc)); }
}
__global__ void k_gat2_p2(const int* __restrict__ d1, const int* __restrict__ v1,
                          float* __restrict__ es, const unsigned* __restrict__ mkey,
                          float* __restrict__ den, int E, int Kn){
  int e = blockIdx.x*blockDim.x + threadIdx.x;
  if (e >= E + Kn) return;
  int d;
  if (e < E){ if (!v1[e]) return; d = d1[e]; } else d = e - E;
  float ex = __expf(es[e] - funkey(mkey[d]));
  es[e] = ex;
  atomicAdd(&den[d], ex);
}
__global__ void k_gat2_p3(const int* __restrict__ s1, const int* __restrict__ d1, const int* __restrict__ v1,
                          const float* __restrict__ xl, const float* __restrict__ es,
                          const float* __restrict__ den, float* __restrict__ agg, int E, int Kn){
  int gid = blockIdx.x*blockDim.x + threadIdx.x;
  int e = gid >> 5, lane = gid & 31;
  if (e >= E + Kn) return;
  int s, d;
  if (e < E){ if (!v1[e]) return; s = s1[e]; d = d1[e]; } else { s = d = e - E; }
  float a = es[e] / den[d];
  atomicAdd(&agg[(size_t)d*HID + lane*2    ], a * xl[(size_t)s*HID + lane*2    ]);
  atomicAdd(&agg[(size_t)d*HID + lane*2 + 1], a * xl[(size_t)s*HID + lane*2 + 1]);
}

// ---------------- misc node kernels ----------------
__global__ void k_bias_relu(float* __restrict__ x, const float* __restrict__ b, long n, int dcols){
  long i = (long)blockIdx.x*blockDim.x + threadIdx.x;
  if (i < n){ float v = x[i] + b[i % dcols]; x[i] = v > 0.f ? v : 0.f; }
}
// xl1[n] = X[n,:] . w ; al = xl1*asrc ; ar = xl1*adst   (wave per node)
__global__ void k_node_score_proj(const float* __restrict__ X, int n, int dcols, const float* __restrict__ w,
                                  const float* __restrict__ asrc, const float* __restrict__ adst,
                                  float* __restrict__ xl1, float* __restrict__ al, float* __restrict__ ar){
  int gid = blockIdx.x*blockDim.x + threadIdx.x;
  int node = gid >> 5, lane = gid & 31;
  if (node >= n) return;
  const float* px = X + (size_t)node*dcols;
  float acc = 0.f;
  for (int c = lane; c < dcols; c += 32) acc += px[c]*w[c];
  #pragma unroll
  for (int o = 16; o > 0; o >>= 1) acc += __shfl_xor(acc, o, 32);
  if (lane == 0){ xl1[node] = acc; al[node] = acc*asrc[0]; ar[node] = acc*adst[0]; }
}

// ---------------- pool score GNN (scalar, 1 head): thread per edge ----------------
__global__ void k_ps_p1(const int* __restrict__ src, const int* __restrict__ dst, const int* __restrict__ valid,
                        const float* __restrict__ al, const float* __restrict__ ar,
                        float* __restrict__ es, unsigned* __restrict__ mkey, int E, int Nn){
  int e = blockIdx.x*blockDim.x + threadIdx.x;
  if (e >= E + Nn) return;
  int s, d;
  if (e < E){ if (valid && !valid[e]) return; s = src[e]; d = dst[e]; } else { s = d = e - E; }
  float t = lrelu(al[s] + ar[d]);
  es[e] = t;
  atomicMax(&mkey[d], fkey(t));
}
__global__ void k_ps_p2(const int* __restrict__ dst, const int* __restrict__ valid,
                        float* __restrict__ es, const unsigned* __restrict__ mkey,
                        float* __restrict__ den, int E, int Nn){
  int e = blockIdx.x*blockDim.x + threadIdx.x;
  if (e >= E + Nn) return;
  int d;
  if (e < E){ if (valid && !valid[e]) return; d = dst[e]; } else d = e - E;
  float ex = __expf(es[e] - funkey(mkey[d]));
  es[e] = ex;
  atomicAdd(&den[d], ex);
}
__global__ void k_ps_p3(const int* __restrict__ src, const int* __restrict__ dst, const int* __restrict__ valid,
                        const float* __restrict__ es, const float* __restrict__ den,
                        const float* __restrict__ xl1, float* __restrict__ attn, int E, int Nn){
  int e = blockIdx.x*blockDim.x + threadIdx.x;
  if (e >= E + Nn) return;
  int s, d;
  if (e < E){ if (valid && !valid[e]) return; s = src[e]; d = dst[e]; } else { s = d = e - E; }
  atomicAdd(&attn[d], es[e] / den[d] * xl1[s]);
}
__global__ void k_score_tanh(const float* __restrict__ attn, const float* __restrict__ b,
                             const float* __restrict__ sel, float* __restrict__ score,
                             unsigned* __restrict__ keys, int n){
  int i = blockIdx.x*blockDim.x + threadIdx.x;
  if (i >= n) return;
  float s = tanhf((attn[i] + b[0]) * sel[0] / fabsf(sel[0]));
  score[i] = s;
  keys[i]  = fkey(s);
}

// ---------------- top-k radix select (state layout: [0..255]=hist, 256=prefix, 257=kneed,
//                  258=count_gt, 259=ctr, 260=ctrEq, 261=eqneed) ----------------
__global__ void k_sel_init(unsigned* st, int k){ if (threadIdx.x == 0){ st[256] = 0u; st[257] = (unsigned)k; } }
__global__ void k_sel_hist(const unsigned* __restrict__ keys, int n, unsigned* __restrict__ st,
                           int shift, unsigned himask){
  int i = blockIdx.x*blockDim.x + threadIdx.x;
  if (i >= n) return;
  unsigned k = keys[i];
  if ((k & himask) != st[256]*0u + (st[256] & himask)) return;
  atomicAdd(&st[(k >> shift) & 0xFFu], 1u);
}
__global__ void k_sel_pick(unsigned* st, int shift){
  if (threadIdx.x != 0) return;
  unsigned kneed = st[257], cum = 0;
  for (int b = 255; b >= 0; --b){
    unsigned c = st[b];
    if (cum + c >= kneed){ st[256] |= ((unsigned)b) << shift; st[257] = kneed - cum; return; }
    cum += c;
  }
  st[257] = kneed - cum;  // fallback (should not happen)
}
__global__ void k_sel_count_gt(const unsigned* __restrict__ keys, int n, unsigned* __restrict__ st){
  int i = blockIdx.x*blockDim.x + threadIdx.x;
  if (i < n && keys[i] > st[256]) atomicAdd(&st[258], 1u);
}
__global__ void k_sel_fin(unsigned* st, int k){ if (threadIdx.x == 0) st[261] = (unsigned)k - st[258]; }
__global__ void k_sel_compact(const unsigned* __restrict__ keys, const float* __restrict__ score, int n,
                              unsigned* __restrict__ st, int k, int* __restrict__ perm,
                              float* __restrict__ vals, int* __restrict__ newid){
  int i = blockIdx.x*blockDim.x + threadIdx.x;
  if (i >= n) return;
  unsigned key = keys[i], T = st[256];
  int pos = -1;
  if (key > T) pos = (int)atomicAdd(&st[259], 1u);
  else if (key == T){
    unsigned t = atomicAdd(&st[260], 1u);
    if (t < st[261]) pos = (int)atomicAdd(&st[259], 1u);
  }
  if (pos >= 0 && pos < k){
    perm[pos] = i;
    vals[pos] = score[i];
    if (newid) newid[i] = pos;
  }
}

// ---------------- pooled feature gather, edge remap, column reductions ----------------
__global__ void k_pool_gather(const float* __restrict__ X, const int* __restrict__ perm,
                              const float* __restrict__ vals, float* __restrict__ out, int k, int dcols){
  long i = (long)blockIdx.x*blockDim.x + threadIdx.x;
  if (i < (long)k*dcols){
    int r = (int)(i / dcols), c = (int)(i % dcols);
    out[i] = X[(size_t)perm[r]*dcols + c] * vals[r];
  }
}
__global__ void k_edge_remap(const int* __restrict__ src, const int* __restrict__ dst,
                             const int* __restrict__ newid, int* __restrict__ s1,
                             int* __restrict__ d1, int* __restrict__ v1, int E){
  int e = blockIdx.x*blockDim.x + threadIdx.x;
  if (e >= E) return;
  int a = newid[src[e]], b = newid[dst[e]];
  int v = (a >= 0) && (b >= 0);
  s1[e] = v ? a : 0; d1[e] = v ? b : 0; v1[e] = v;
}
// partial column max/sum over row-chunks (optionally gathered/scaled via perm/vals)
__global__ void k_colred_part(const float* __restrict__ X, const int* __restrict__ perm,
                              const float* __restrict__ vals, int rows, int cols,
                              float* __restrict__ pmax, float* __restrict__ psum, int nchunks){
  int c = blockIdx.x*blockDim.x + threadIdx.x;
  int chunk = blockIdx.y;
  if (c >= cols) return;
  int r0 = (int)(((long)rows *  chunk   ) / nchunks);
  int r1 = (int)(((long)rows * (chunk+1)) / nchunks);
  float mx = -3.0e38f, sm = 0.f;
  for (int r = r0; r < r1; ++r){
    int rr = perm ? perm[r] : r;
    float v = X[(size_t)rr*cols + c];
    if (vals) v *= vals[r];
    mx = fmaxf(mx, v); sm += v;
  }
  pmax[(size_t)chunk*cols + c] = mx;
  psum[(size_t)chunk*cols + c] = sm;
}
__global__ void k_colred_fin(const float* __restrict__ pmax, const float* __restrict__ psum,
                             int cols, int nchunks, int rows,
                             float* __restrict__ omax, float* __restrict__ omean){
  int c = blockIdx.x*blockDim.x + threadIdx.x;
  if (c >= cols) return;
  float mx = -3.0e38f, sm = 0.f;
  for (int k = 0; k < nchunks; ++k){ mx = fmaxf(mx, pmax[(size_t)k*cols + c]); sm += psum[(size_t)k*cols + c]; }
  omax[c] = mx; omean[c] = sm / rows;
}

// ---------------- head ----------------
__global__ void k_build_z(const float* __restrict__ x1max, const float* __restrict__ x1mean,
                          const float* __restrict__ m2max, const float* __restrict__ m2mean,
                          float* __restrict__ z){
  int i = blockIdx.x*blockDim.x + threadIdx.x;
  if (i >= 1024) return;
  float a = (i < 512) ? x1max[i] : x1mean[i - 512];
  float g = (i < 512) ? m2max[i & 63] : m2mean[i & 63];
  z[i] = a + g;
}
__global__ void k_gemv(const float* __restrict__ in, const float* __restrict__ w,
                       const float* __restrict__ b, float* __restrict__ out,
                       int din, int dout, int relu){
  int j = blockIdx.x*blockDim.x + threadIdx.x;
  if (j >= dout) return;
  float acc = b[j];
  for (int i = 0; i < din; ++i) acc += in[i] * w[(size_t)i*dout + j];
  out[j] = (relu && acc < 0.f) ? 0.f : acc;
}
__global__ void k_final(const float* __restrict__ logits, float* __restrict__ out){
  if (threadIdx.x != 0) return;
  float a = logits[0], b = logits[1];
  float m = fmaxf(a, b);
  float ea = __expf(a - m), eb = __expf(b - m), s = ea + eb;
  out[0] = a; out[1] = b; out[2] = ea/s; out[3] = eb/s;
}

// ---------------- host sequencing ----------------
extern "C" void kernel_launch(void* const* d_in, const int* in_sizes, int n_in,
                              void* d_out, int out_size, void* d_ws, size_t ws_size,
                              hipStream_t stream) {
  (void)in_sizes; (void)n_in; (void)out_size; (void)ws_size;
  auto cdiv = [](long a, long b){ return (unsigned)((a + b - 1) / b); };
  char* ws = (char*)d_ws;

  // inputs
  const float* x      = (const float*)d_in[0];
  const int*   ei     = (const int*)  d_in[1];
  const int*   src    = ei;
  const int*   dst    = ei + E_;
  const float* g1_wl  = (const float*)d_in[2];
  const float* g1_wr  = (const float*)d_in[3];
  const float* g1_att = (const float*)d_in[4];
  const float* g1_b   = (const float*)d_in[5];
  const float* p1_w   = (const float*)d_in[6];
  const float* p1_as  = (const float*)d_in[7];
  const float* p1_ad  = (const float*)d_in[8];
  const float* p1_b   = (const float*)d_in[9];
  const float* p1_sel = (const float*)d_in[10];
  const float* g2_wl  = (const float*)d_in[11];
  const float* g2_wr  = (const float*)d_in[12];
  const float* g2_att = (const float*)d_in[13];
  const float* g2_b   = (const float*)d_in[14];
  const float* p2_w   = (const float*)d_in[15];
  const float* p2_as  = (const float*)d_in[16];
  const float* p2_ad  = (const float*)d_in[17];
  const float* p2_b   = (const float*)d_in[18];
  const float* p2_sel = (const float*)d_in[19];
  const float* l1_w   = (const float*)d_in[20];
  const float* l1_b   = (const float*)d_in[21];
  const float* l2_w   = (const float*)d_in[22];
  const float* l2_b   = (const float*)d_in[23];
  const float* l3_w   = (const float*)d_in[24];
  const float* l3_b   = (const float*)d_in[25];
  const float* l4_w   = (const float*)d_in[26];
  const float* l4_b   = (const float*)d_in[27];

  // workspace views
  __bf16* xb    = (__bf16*)(ws + OFF_XB);
  __bf16* w1lT  = (__bf16*)(ws + OFF_W1T);
  __bf16* w1rT  = w1lT + (size_t)HH*DIN;
  float*  xl1   = (float*)(ws + OFF_XL1);
  float*  xr1   = (float*)(ws + OFF_XR1);
  float*  agg1  = (float*)(ws + OFF_AGG1);
  float*  esc   = (float*)(ws + OFF_ESC);
  unsigned* mkey= (unsigned*)(ws + OFF_MKEY);
  float*  den   = (float*)(ws + OFF_DEN);
  float*  nxl   = (float*)(ws + OFF_NODE);            // per-node scalar projection
  float*  nal   = nxl + N_;
  float*  nar   = nal + N_;
  float*  nattn = nar + N_;
  float*  nscore= nattn + N_;
  unsigned* nkeys = (unsigned*)(nscore + N_);
  unsigned* st  = (unsigned*)(ws + OFF_STATE);
  int*    perm  = (int*)(ws + OFF_PERM);
  float*  vals  = (float*)(perm + K1_);
  int*    newid = (int*)(vals + K1_);
  int*    perm2 = newid + N_;
  float*  vals2 = (float*)(perm2 + K2_);
  int*    s1    = (int*)(ws + OFF_S1);
  int*    d1    = s1 + E_;
  int*    v1    = d1 + E_;
  float*  mf    = (float*)(ws + OFF_MISC);
  float*  x1max = mf, *x1mean = mf + 512;
  float*  m2max = mf + 1024, *m2mean = mf + 1088;
  float*  zv    = mf + 1152;
  float*  t1    = mf + 2176, *t2 = mf + 2688, *t3 = mf + 2944, *logitsb = mf + 3072;
  __bf16* xpb   = (__bf16*)(ws + OFF_XPB);
  float*  xl2   = (float*)(ws + OFF_XL2);
  float*  xr2   = (float*)(ws + OFF_XR2);
  float*  agg2  = (float*)(ws + OFF_AGG2);
  __bf16* w2lT  = (__bf16*)(ws + OFF_W2T);
  __bf16* w2rT  = w2lT + (size_t)HID*HH;
  float*  xp    = (float*)(ws + OFF_XP);
  float*  pmax  = esc;                // colreduce scratch (ESC free in its windows)
  float*  psum  = pmax + 64*512;

  const unsigned B = 256;

  // ===== Stage A: gat1 feature transforms (bf16 WMMA) =====
  k_f32_to_bf16 <<<cdiv((long)N_*DIN, B), B, 0, stream>>>(x, xb, (long)N_*DIN);
  k_transpose_bf16<<<cdiv((long)DIN*HH, B), B, 0, stream>>>(g1_wl, w1lT, DIN, HH);
  k_transpose_bf16<<<cdiv((long)DIN*HH, B), B, 0, stream>>>(g1_wr, w1rT, DIN, HH);
  {
    long tiles = (long)((N_+15)/16) * (HH/16);
    k_wmma_gemm<<<cdiv(tiles, 8), B, 0, stream>>>(xb, w1lT, xl1, N_, HH, DIN);
    k_wmma_gemm<<<cdiv(tiles, 8), B, 0, stream>>>(xb, w1rT, xr1, N_, HH, DIN);
  }
  // ===== gat1 edge softmax + aggregate =====
  hipMemsetAsync(mkey, 0, (size_t)N_*8*4, stream);
  hipMemsetAsync(den,  0, (size_t)N_*8*4, stream);
  hipMemsetAsync(agg1, 0, (size_t)N_*HH*4, stream);
  k_gat1_p1<<<cdiv((long)E1_*32, B), B, 0, stream>>>(src, dst, xl1, xr1, g1_att, esc, mkey, E_, N_);
  k_gat1_p2<<<cdiv((long)E1_, B), B, 0, stream>>>(dst, esc, mkey, den, E_, N_);
  k_gat1_p3<<<cdiv((long)E1_*32, B), B, 0, stream>>>(src, dst, xl1, esc, den, agg1, E_, N_);
  k_bias_relu<<<cdiv((long)N_*HH, B), B, 0, stream>>>(agg1, g1_b, (long)N_*HH, HH);  // h1

  // ===== Stage B: pool1 =====
  k_node_score_proj<<<cdiv((long)N_*32, B), B, 0, stream>>>(agg1, N_, HH, p1_w, p1_as, p1_ad, nxl, nal, nar);
  hipMemsetAsync(mkey, 0, (size_t)N_*4, stream);
  hipMemsetAsync(den,  0, (size_t)N_*4, stream);
  hipMemsetAsync(nattn,0, (size_t)N_*4, stream);
  k_ps_p1<<<cdiv((long)E1_, B), B, 0, stream>>>(src, dst, nullptr, nal, nar, esc, mkey, E_, N_);
  k_ps_p2<<<cdiv((long)E1_, B), B, 0, stream>>>(dst, nullptr, esc, mkey, den, E_, N_);
  k_ps_p3<<<cdiv((long)E1_, B), B, 0, stream>>>(src, dst, nullptr, esc, den, nxl, nattn, E_, N_);
  k_score_tanh<<<cdiv((long)N_, B), B, 0, stream>>>(nattn, p1_b, p1_sel, nscore, nkeys, N_);
  // top-K1 radix select
  k_sel_init<<<1, 32, 0, stream>>>(st, K1_);
  const int shifts[4] = {24, 16, 8, 0};
  const unsigned himasks[4] = {0u, 0xFF000000u, 0xFFFF0000u, 0xFFFFFF00u};
  for (int p = 0; p < 4; ++p){
    hipMemsetAsync(st, 0, 1024, stream);
    k_sel_hist<<<cdiv((long)N_, B), B, 0, stream>>>(nkeys, N_, st, shifts[p], himasks[p]);
    k_sel_pick<<<1, 32, 0, stream>>>(st, shifts[p]);
  }
  hipMemsetAsync(st + 258, 0, 12, stream);
  k_sel_count_gt<<<cdiv((long)N_, B), B, 0, stream>>>(nkeys, N_, st);
  k_sel_fin<<<1, 32, 0, stream>>>(st, K1_);
  hipMemsetAsync(newid, 0xFF, (size_t)N_*4, stream);   // -1
  k_sel_compact<<<cdiv((long)N_, B), B, 0, stream>>>(nkeys, nscore, N_, st, K1_, perm, vals, newid);
  k_pool_gather<<<cdiv((long)K1_*HH, B), B, 0, stream>>>(agg1, perm, vals, xp, K1_, HH);
  k_edge_remap<<<cdiv((long)E_, B), B, 0, stream>>>(src, dst, newid, s1, d1, v1, E_);
  // x1 = [max(xp), mean(xp)]  (ESC region free here -> scratch)
  k_colred_part<<<dim3(cdiv(HH, B), 64), B, 0, stream>>>(xp, nullptr, nullptr, K1_, HH, pmax, psum, 64);
  k_colred_fin <<<cdiv(HH, B), B, 0, stream>>>(pmax, psum, HH, 64, K1_, x1max, x1mean);

  // ===== Stage D: gat2 on pooled graph (bf16 WMMA) =====
  k_f32_to_bf16 <<<cdiv((long)K1_*HH, B), B, 0, stream>>>(xp, xpb, (long)K1_*HH);
  k_transpose_bf16<<<cdiv((long)HH*HID, B), B, 0, stream>>>(g2_wl, w2lT, HH, HID);
  k_transpose_bf16<<<cdiv((long)HH*HID, B), B, 0, stream>>>(g2_wr, w2rT, HH, HID);
  {
    long tiles = (long)((K1_+15)/16) * (HID/16);
    k_wmma_gemm<<<cdiv(tiles, 8), B, 0, stream>>>(xpb, w2lT, xl2, K1_, HID, HH);
    k_wmma_gemm<<<cdiv(tiles, 8), B, 0, stream>>>(xpb, w2rT, xr2, K1_, HID, HH);
  }
  hipMemsetAsync(mkey, 0, (size_t)K1_*4, stream);
  hipMemsetAsync(den,  0, (size_t)K1_*4, stream);
  hipMemsetAsync(agg2, 0, (size_t)K1_*HID*4, stream);
  k_gat2_p1<<<cdiv((long)E2_*32, B), B, 0, stream>>>(s1, d1, v1, xl2, xr2, g2_att, esc, mkey, E_, K1_);
  k_gat2_p2<<<cdiv((long)E2_, B), B, 0, stream>>>(d1, v1, esc, mkey, den, E_, K1_);
  k_gat2_p3<<<cdiv((long)E2_*32, B), B, 0, stream>>>(s1, d1, v1, xl2, esc, den, agg2, E_, K1_);
  k_bias_relu<<<cdiv((long)K1_*HID, B), B, 0, stream>>>(agg2, g2_b, (long)K1_*HID, HID);  // h2

  // ===== Stage E: pool2 =====
  k_node_score_proj<<<cdiv((long)K1_*32, B), B, 0, stream>>>(agg2, K1_, HID, p2_w, p2_as, p2_ad, nxl, nal, nar);
  hipMemsetAsync(mkey, 0, (size_t)K1_*4, stream);
  hipMemsetAsync(den,  0, (size_t)K1_*4, stream);
  hipMemsetAsync(nattn,0, (size_t)K1_*4, stream);
  k_ps_p1<<<cdiv((long)E2_, B), B, 0, stream>>>(s1, d1, v1, nal, nar, esc, mkey, E_, K1_);
  k_ps_p2<<<cdiv((long)E2_, B), B, 0, stream>>>(d1, v1, esc, mkey, den, E_, K1_);
  k_ps_p3<<<cdiv((long)E2_, B), B, 0, stream>>>(s1, d1, v1, esc, den, nxl, nattn, E_, K1_);
  k_score_tanh<<<cdiv((long)K1_, B), B, 0, stream>>>(nattn, p2_b, p2_sel, nscore, nkeys, K1_);
  k_sel_init<<<1, 32, 0, stream>>>(st, K2_);
  for (int p = 0; p < 4; ++p){
    hipMemsetAsync(st, 0, 1024, stream);
    k_sel_hist<<<cdiv((long)K1_, B), B, 0, stream>>>(nkeys, K1_, st, shifts[p], himasks[p]);
    k_sel_pick<<<1, 32, 0, stream>>>(st, shifts[p]);
  }
  hipMemsetAsync(st + 258, 0, 12, stream);
  k_sel_count_gt<<<cdiv((long)K1_, B), B, 0, stream>>>(nkeys, K1_, st);
  k_sel_fin<<<1, 32, 0, stream>>>(st, K2_);
  k_sel_compact<<<cdiv((long)K1_, B), B, 0, stream>>>(nkeys, nscore, K1_, st, K2_, perm2, vals2, nullptr);
  // global max/mean over scaled selected h2 rows (ESC free again -> scratch)
  k_colred_part<<<dim3(cdiv(HID, B), 64), B, 0, stream>>>(agg2, perm2, vals2, K2_, HID, pmax, psum, 64);
  k_colred_fin <<<cdiv(HID, B), B, 0, stream>>>(pmax, psum, HID, 64, K2_, m2max, m2mean);

  // ===== Stage F: MLP head =====
  k_build_z<<<4, B, 0, stream>>>(x1max, x1mean, m2max, m2mean, zv);
  k_gemv<<<cdiv(512, B), B, 0, stream>>>(zv, l1_w, l1_b, t1, 1024, 512, 1);
  k_gemv<<<cdiv(256, B), B, 0, stream>>>(t1, l2_w, l2_b, t2, 512, 256, 1);
  k_gemv<<<cdiv(128, B), B, 0, stream>>>(t2, l3_w, l3_b, t3, 256, 128, 1);
  k_gemv<<<1, B, 0, stream>>>(t3, l4_w, l4_b, logitsb, 128, 2, 0);
  k_final<<<1, 32, 0, stream>>>(logitsb, (float*)d_out);
}
